// MQA_4578435137944
// MI455X (gfx1250) — matrix operand order repaired and checked by reading, not compiled, not run on hardware
//
#include <hip/hip_runtime.h>
#include <hip/hip_bf16.h>

// ---------------------------------------------------------------------------
// MQA forward for gfx1250 (wave32, WMMA bf16 16x16x32, f32 accumulate).
// B=4, T=2048, D=1024, HS=128, NKV(heads)=2.
// ---------------------------------------------------------------------------

typedef __attribute__((ext_vector_type(16))) __bf16 v16bf;
typedef __attribute__((ext_vector_type(8)))  float  v8f;

#define WMMA_BF16(A, B, C) \
  __builtin_amdgcn_wmma_f32_16x16x32_bf16(false, (A), false, (B), (short)0, (C), false, false)

// ---- Fragment loaders (A and B share the contiguous k order K = 16h+e) ----------

// A matrix 16x32 (MxK), row-major source with stride lda (elements).
// lane = m + 16*h ; elem e -> K = 16h+e
template <typename SrcT>
__device__ inline v16bf load_A(const SrcT* __restrict__ p, int lda) {
  const int lane = threadIdx.x & 31;
  const int h = lane >> 4, m = lane & 15;
  v16bf a;
#pragma unroll
  for (int e = 0; e < 8; ++e) {
    a[e]     = (__bf16)(float)p[(size_t)m * lda + 16 * h + e];
    a[8 + e] = (__bf16)(float)p[(size_t)m * lda + 16 * h + 8 + e];
  }
  return a;
}

// B matrix 32x16 (KxN) where B[k][n] = src[n*ld + k]  (i.e. source holds N rows
// of length K: weight matrices W[out,in], or the key matrix K[s,d]).
// lane = n + 16*h ; elem e -> K = 16h+e
template <typename SrcT>
__device__ inline v16bf load_B_T(const SrcT* __restrict__ p, int ld) {
  const int lane = threadIdx.x & 31;
  const int h = lane >> 4, n = lane & 15;
  v16bf b;
#pragma unroll
  for (int e = 0; e < 16; ++e)
    b[e] = (__bf16)(float)p[(size_t)n * ld + 16 * h + e];
  return b;
}

// B matrix 32x16 (KxN) where B[k][n] = src[k*ld + n]  (row-major KxN: V matrix).
template <typename SrcT>
__device__ inline v16bf load_B_N(const SrcT* __restrict__ p, int ld) {
  const int lane = threadIdx.x & 31;
  const int h = lane >> 4, n = lane & 15;
  v16bf b;
#pragma unroll
  for (int e = 0; e < 16; ++e)
    b[e] = (__bf16)(float)p[(size_t)(16 * h + e) * ld + n];
  return b;
}

// C/D 16x16 f32: register r -> row (r + 8h), col n = lane&15.
__device__ inline void store_C_bf16(v8f c, __bf16* __restrict__ p, int ld) {
  const int lane = threadIdx.x & 31;
  const int h = lane >> 4, n = lane & 15;
#pragma unroll
  for (int r = 0; r < 8; ++r)
    p[(size_t)(r + 8 * h) * ld + n] = (__bf16)c[r];
}

// Row-wise reductions: a C-fragment row lives across the 16 lanes of a half.
__device__ inline float rowmax16(float x) {
#pragma unroll
  for (int m = 8; m >= 1; m >>= 1) x = fmaxf(x, __shfl_xor(x, m, 32));
  return x;
}
__device__ inline float rowsum16(float x) {
#pragma unroll
  for (int m = 8; m >= 1; m >>= 1) x += __shfl_xor(x, m, 32);
  return x;
}

// ---------------------------------------------------------------------------
// Kernel 1: QKV projections + RoPE.  grid = (B*T/16)*4 blocks of 1 wave.
// which: 0,1 -> q head 0/1 (rope); 2 -> k (rope); 3 -> v.
// ---------------------------------------------------------------------------
__global__ void mqa_proj_rope(const float* __restrict__ x,
                              const float* __restrict__ Wq,
                              const float* __restrict__ Wk,
                              const float* __restrict__ Wv,
                              __bf16* __restrict__ q_ws,
                              __bf16* __restrict__ k_ws,
                              __bf16* __restrict__ v_ws) {
  const int D = 1024, HS = 128, T = 2048, NKV = 2;
  const int rt = blockIdx.x >> 2;
  const int which = blockIdx.x & 3;
  const float* W = (which == 0) ? Wq
                 : (which == 1) ? (Wq + (size_t)HS * D)
                 : (which == 2) ? Wk : Wv;
  const int row0 = rt * 16;          // global row in [0, B*T)
  const int b = row0 / T;
  const int t0 = row0 - b * T;       // position within sequence (tile-aligned)
  const int lane = threadIdx.x & 31;
  const int h = lane >> 4, n = lane & 15;

  __bf16* dst;
  if (which < 2)      dst = q_ws + ((size_t)(b * NKV + which) * T + t0) * HS;
  else if (which == 2) dst = k_ws + ((size_t)b * T + t0) * HS;
  else                 dst = v_ws + ((size_t)b * T + t0) * HS;

  const v8f vzero = {0.f, 0.f, 0.f, 0.f, 0.f, 0.f, 0.f, 0.f};

  for (int pass = 0; pass < 2; ++pass) {          // 2 x 4 N-tiles = 128 cols
    v8f acc[4];
#pragma unroll
    for (int j = 0; j < 4; ++j) acc[j] = vzero;

    for (int kk = 0; kk < 32; ++kk) {             // K = 1024 / 32
      v16bf a = load_A<float>(x + (size_t)row0 * D + kk * 32, D);
#pragma unroll
      for (int j = 0; j < 4; ++j) {
        v16bf bf = load_B_T<float>(W + (size_t)(pass * 64 + j * 16) * D + kk * 32, D);
        acc[j] = WMMA_BF16(a, bf, acc[j]);
      }
    }

#pragma unroll
    for (int j = 0; j < 4; ++j) {
      const int cb = pass * 64 + j * 16;
      if (which != 3) {                            // RoPE on q and k
#pragma unroll
        for (int r = 0; r < 8; ++r) {
          float v0 = acc[j][r];
          float vp = __shfl_xor(v0, 1, 32);        // adjacent column = adjacent lane
          int c = cb + n;
          int jp = c >> 1;
          // theta_j = 10000^(-j/32) = exp(-j * ln(10000)/32)
          float theta = __expf(-(float)jp * 0.2878231366242557f);
          float ang = (float)(t0 + r + 8 * h) * theta;
          float cs, sn;
          sincosf(ang, &sn, &cs);                  // sn=sin, cs=cos
          acc[j][r] = (c & 1) ? (v0 * cs + vp * sn) : (v0 * cs - vp * sn);
        }
      }
      store_C_bf16(acc[j], dst + cb, HS);
    }
  }
}

// ---------------------------------------------------------------------------
// Kernel 2: causal flash attention. 1 wave per 16-row Q tile, 32-key blocks.
// grid = 256 blocks x 128 threads (4 independent waves per block).
// ---------------------------------------------------------------------------
__global__ void mqa_flash_attn(const __bf16* __restrict__ q_ws,
                               const __bf16* __restrict__ k_ws,
                               const __bf16* __restrict__ v_ws,
                               __bf16* __restrict__ attn_ws) {
  const int T = 2048, HS = 128, NKV = 2;
  __shared__ __align__(16) __bf16 pst[4][16][32];   // per-wave P staging (C->A transpose)

  const int wid = threadIdx.x >> 5;
  const int wg = blockIdx.x * 4 + wid;              // 0..1023
  const int b = wg >> 8;
  const int rem = wg & 255;
  const int hh = rem >> 7;
  const int tile = rem & 127;
  const int ql = tile * 16;

  const __bf16* Q = q_ws + (((size_t)b * NKV + hh) * T + ql) * HS;
  const __bf16* K = k_ws + (size_t)b * T * HS;
  const __bf16* V = v_ws + (size_t)b * T * HS;

  const int lane = threadIdx.x & 31;
  const int h = lane >> 4, n = lane & 15;
  const float scale = 0.08838834764831845f;         // 1/sqrt(128)
  const v8f vzero = {0.f, 0.f, 0.f, 0.f, 0.f, 0.f, 0.f, 0.f};

  v16bf qf[4];                                      // Q tile 16x128 (4 K-chunks)
#pragma unroll
  for (int s = 0; s < 4; ++s) qf[s] = load_A<__bf16>(Q + s * 32, HS);

  v8f acc[8];                                       // O tile 16x128
#pragma unroll
  for (int nt = 0; nt < 8; ++nt) acc[nt] = vzero;
  float mrun[8], lrun[8];
#pragma unroll
  for (int r = 0; r < 8; ++r) { mrun[r] = -1e30f; lrun[r] = 0.f; }

  const int nblk = (ql + 47) >> 5;                  // causal: ((ql+15)/32)+1

  for (int kb = 0; kb < nblk; ++kb) {
    const int kbase = kb * 32;
    v8f s0 = vzero, s1 = vzero;                     // S tile 16x32 (two 16x16)
#pragma unroll
    for (int ks = 0; ks < 4; ++ks) {
      v16bf b0 = load_B_T<__bf16>(K + (size_t)kbase * HS + ks * 32, HS);
      s0 = WMMA_BF16(qf[ks], b0, s0);
      v16bf b1 = load_B_T<__bf16>(K + (size_t)(kbase + 16) * HS + ks * 32, HS);
      s1 = WMMA_BF16(qf[ks], b1, s1);
    }

    // online softmax (rows live across 16-lane halves at fixed reg index r)
#pragma unroll
    for (int r = 0; r < 8; ++r) {
      const int i = ql + r + 8 * h;
      float a0 = (kbase + n      <= i) ? s0[r] * scale : -1e30f;
      float a1 = (kbase + 16 + n <= i) ? s1[r] * scale : -1e30f;
      float mb = fmaxf(rowmax16(fmaxf(a0, a1)), mrun[r]);
      float corr = __expf(mrun[r] - mb);
      float e0 = __expf(a0 - mb);
      float e1 = __expf(a1 - mb);
      lrun[r] = lrun[r] * corr + rowsum16(e0 + e1);
      mrun[r] = mb;
#pragma unroll
      for (int nt = 0; nt < 8; ++nt) acc[nt][r] *= corr;
      pst[wid][r + 8 * h][n]      = (__bf16)e0;     // C-layout -> LDS
      pst[wid][r + 8 * h][16 + n] = (__bf16)e1;
    }
    asm volatile("s_wait_dscnt 0" ::: "memory");     // stores visible before A-load

    v16bf pA = load_A<__bf16>(&pst[wid][0][0], 32);  // P in A-layout (16x32)
#pragma unroll
    for (int nt = 0; nt < 8; ++nt) {
      v16bf bv = load_B_N<__bf16>(V + (size_t)kbase * HS + nt * 16, HS);
      acc[nt] = WMMA_BF16(pA, bv, acc[nt]);
    }
  }

  // normalize and store to attn_ws[b*T+t][hh*128 + c]  (concat-heads layout)
  __bf16* dst = attn_ws + ((size_t)b * T + ql) * (NKV * HS) + hh * HS;
#pragma unroll
  for (int nt = 0; nt < 8; ++nt) {
    v8f o;
#pragma unroll
    for (int r = 0; r < 8; ++r) o[r] = acc[nt][r] / lrun[r];
    store_C_bf16(o, dst + nt * 16, NKV * HS);
  }
}

// ---------------------------------------------------------------------------
// Kernel 3: output projection  attn[B*T,256] @ Wo^T -> out[B*T,1024] (f32).
// grid = (B*T/16) * 16 blocks of 1 wave; each wave does 4 N-tiles (64 cols).
// ---------------------------------------------------------------------------
__global__ void mqa_outproj(const __bf16* __restrict__ attn_ws,
                            const float* __restrict__ Wo,
                            float* __restrict__ out) {
  const int D = 1024, C = 256;
  const int rt = blockIdx.x >> 4;
  const int ng = blockIdx.x & 15;
  const int lane = threadIdx.x & 31;
  const int h = lane >> 4, n = lane & 15;
  const v8f vzero = {0.f, 0.f, 0.f, 0.f, 0.f, 0.f, 0.f, 0.f};

  v8f acc[4];
#pragma unroll
  for (int j = 0; j < 4; ++j) acc[j] = vzero;

  for (int kk = 0; kk < 8; ++kk) {                  // K = 256 / 32
    v16bf a = load_A<__bf16>(attn_ws + (size_t)rt * 16 * C + kk * 32, C);
#pragma unroll
    for (int j = 0; j < 4; ++j) {
      v16bf bf = load_B_T<float>(Wo + (size_t)(ng * 64 + j * 16) * C + kk * 32, C);
      acc[j] = WMMA_BF16(a, bf, acc[j]);
    }
  }

#pragma unroll
  for (int j = 0; j < 4; ++j)
#pragma unroll
    for (int r = 0; r < 8; ++r)
      out[(size_t)(rt * 16 + r + 8 * h) * D + ng * 64 + j * 16 + n] = acc[j][r];
}

// ---------------------------------------------------------------------------
extern "C" void kernel_launch(void* const* d_in, const int* in_sizes, int n_in,
                              void* d_out, int out_size, void* d_ws, size_t ws_size,
                              hipStream_t stream) {
  const float* x  = (const float*)d_in[0];   // [4,2048,1024]
  const float* Wq = (const float*)d_in[1];   // [2,128,1024]
  const float* Wk = (const float*)d_in[2];   // [128,1024]
  const float* Wv = (const float*)d_in[3];   // [128,1024]
  const float* Wo = (const float*)d_in[4];   // [1024,256]
  float* out = (float*)d_out;                // [4,2048,1024]

  const size_t B = 4, T = 2048, HS = 128, NKV = 2;
  __bf16* q_ws = (__bf16*)d_ws;                          // [B,NKV,T,HS]
  __bf16* k_ws = q_ws + B * NKV * T * HS;                // [B,T,HS]
  __bf16* v_ws = k_ws + B * T * HS;                      // [B,T,HS]
  __bf16* attn_ws = v_ws + B * T * HS;                   // [B,T,NKV*HS]

  // 1) QKV projections + RoPE: 512 row-tiles x 4 outputs, 1 wave each.
  mqa_proj_rope<<<2048, 32, 0, stream>>>(x, Wq, Wk, Wv, q_ws, k_ws, v_ws);
  // 2) causal flash attention: 1024 Q-tiles, 4 waves per block.
  mqa_flash_attn<<<256, 128, 0, stream>>>(q_ws, k_ws, v_ws, attn_ws);
  // 3) output projection: 512 row-tiles x 16 column groups.
  mqa_outproj<<<8192, 32, 0, stream>>>(attn_ws, Wo, out);
}
